// TorchModel_75213467287821
// MI455X (gfx1250) — matrix operand order repaired and checked by reading, not compiled, not run on hardware
//
#include <hip/hip_runtime.h>
#include <hip/hip_bf16.h>
#include <math.h>
#include <stdint.h>

// ---------------- model dims ----------------
#define BB   16
#define SS   256
#define HH   768
#define LL   12
#define NHEAD 12
#define FFD  3072
#define VV   30522
#define CC   9
#define DH   64
#define BS   4096   // BB*SS
#define ZZ   192    // BB*NHEAD

#define LDS_STRIDE 40   // 32 halves data + 8 halves pad (80B rows; 16B aligned, bank-spread)

typedef __attribute__((ext_vector_type(16))) _Float16 v16h;
typedef __attribute__((ext_vector_type(8)))  _Float16 v8h;
typedef __attribute__((ext_vector_type(8)))  float    v8f;

#define WMMA_F16(a,b,c) __builtin_amdgcn_wmma_f32_16x16x32_f16(false,(a),false,(b),(short)0,(c),false,false)

// ---- WMMA fragment loaders (CDNA5 16-bit lane layouts, ISA 7.12.2) ----
// A 16x32 (MxK) row-major tile at p, leading dim lda (halves).
// lane<16: m=lane, K=0..7 & 16..23 ; lane>=16: m=lane-16, K=8..15 & 24..31
__device__ __forceinline__ v16h load_a_frag(const _Float16* __restrict__ p, int lda) {
  const int l  = threadIdx.x & 31;
  const int m  = l & 15;
  const int kb = (l >> 4) << 3;            // 0 or 8
  const _Float16* r = p + (size_t)m * lda + kb;
  v8h lo = *(const v8h*)(r);               // K = kb .. kb+7
  v8h hi = *(const v8h*)(r + 16);          // K = kb+16 .. kb+23
  v16h a;
#pragma unroll
  for (int i = 0; i < 8; ++i) { a[i] = lo[i]; a[i + 8] = hi[i]; }
  return a;
}

// B 32x16 (KxN) tile, sourced from BT[N][K] row-major at p (= &BT[n0][k0]).
// lane<16: n=lane, K=0..7 & 16..23 ; lane>=16: n=lane-16, K=8..15 & 24..31
__device__ __forceinline__ v16h load_bt_frag(const _Float16* __restrict__ p, int ldb) {
  const int l  = threadIdx.x & 31;
  const int n  = l & 15;
  const int kb = (l >> 4) << 3;            // 0 or 8
  const _Float16* r = p + (size_t)n * ldb + kb;
  v8h lo = *(const v8h*)(r);
  v8h hi = *(const v8h*)(r + 16);
  v16h b;
#pragma unroll
  for (int i = 0; i < 8; ++i) { b[i] = lo[i]; b[i + 8] = hi[i]; }
  return b;
}

__device__ __forceinline__ float gelu_exact(float x) {
  return 0.5f * x * (1.0f + erff(x * 0.70710678118654752440f));
}

// C/D 16x16 f32: lane<16: n=lane, rows 0..7 ; lane>=16: n=lane-16, rows 8..15
template<typename OutT, bool GELU>
__device__ __forceinline__ void store_c_frag(v8f c, OutT* __restrict__ p, int ldc,
                                             const float* __restrict__ bias_tile, float scale) {
  const int l  = threadIdx.x & 31;
  const int n  = l & 15;
  const int mb = (l >> 4) << 3;
  const float bv = bias_tile ? bias_tile[n] : 0.0f;
#pragma unroll
  for (int r = 0; r < 8; ++r) {
    float x = c[r] * scale + bv;
    if (GELU) x = gelu_exact(x);
    p[(size_t)(mb + r) * ldc + n] = (OutT)x;
  }
}

// ---------------- generic TN WMMA GEMM, async-LDS software pipeline ----------------
// C[M,N] = A[M,K](f16,row-major) * BT[N,K](f16,row-major)^T + bias ; optional GELU
// block = 256 threads = 8 waves; block tile 128(M) x 128(N); K-step 32.
// Global->LDS staging via GLOBAL_LOAD_ASYNC_TO_LDS_B128 (ASYNCcnt), 3 LDS buffers,
// 2 K-slabs in flight (ASYNCcnt completes in order; each slab-group = 4 insts/wave,
// so s_wait_asynccnt 4 == "oldest group landed, newer may still fly").
// Wave tile 32x64: 8 v_wmma per K-step from 6 LDS fragments.
template<typename OutT, bool GELU>
__global__ void __launch_bounds__(256)
k_gemm_tn(const _Float16* __restrict__ A, const _Float16* __restrict__ BT,
          const float* __restrict__ bias, OutT* __restrict__ C, int M, int N, int K) {
  __shared__ __align__(16) _Float16 lsA[3][128 * LDS_STRIDE];
  __shared__ __align__(16) _Float16 lsB[3][128 * LDS_STRIDE];
  const int t   = threadIdx.x;
  const int wid = t >> 5;
  const int m0b = blockIdx.y * 128;
  const int n0b = blockIdx.x * 128;
  const int wm  = (wid & 3) * 32;          // wave row offset in block tile
  const int wn  = (wid >> 2) * 64;         // wave col offset in block tile

  const _Float16* Ag = A  + (size_t)m0b * K;
  const _Float16* Bg = BT + (size_t)n0b * K;
  const uint32_t baseA[3] = { (uint32_t)(uintptr_t)&lsA[0][0], (uint32_t)(uintptr_t)&lsA[1][0],
                              (uint32_t)(uintptr_t)&lsA[2][0] };
  const uint32_t baseB[3] = { (uint32_t)(uintptr_t)&lsB[0][0], (uint32_t)(uintptr_t)&lsB[1][0],
                              (uint32_t)(uintptr_t)&lsB[2][0] };

  // async-copy a 128x32 half tile (row-major, ld=K) into LDS at ldsbase.
  // 512 x 16B lane-transfers -> 2 instructions per thread (2 per wave's ASYNCcnt).
  auto issue_tile = [&](const _Float16* gsrc, uint32_t ldsbase, int kk) {
#pragma unroll
    for (int j = 0; j < 2; ++j) {
      const int f   = j * 256 + t;
      const int row = f >> 2, q = f & 3;
      const _Float16* g = gsrc + (size_t)row * K + kk + q * 8;
      const uint32_t  l = ldsbase + (uint32_t)(row * LDS_STRIDE + q * 8) * 2u;
      asm volatile("global_load_async_to_lds_b128 %0, %1, off"
                   :: "v"(l), "v"((unsigned long long)(uintptr_t)g) : "memory");
    }
  };

  v8f acc[2][4] = {};
  const int nk = K >> 5;
  issue_tile(Ag, baseA[0], 0);             // slab 0
  issue_tile(Bg, baseB[0], 0);
  if (nk > 1) {                            // slab 1 (2-deep pipeline)
    issue_tile(Ag, baseA[1], 32);
    issue_tile(Bg, baseB[1], 32);
  }

  int cur = 0;
  for (int ks = 0; ks < nk; ++ks) {
    if (ks + 1 < nk) { asm volatile("s_wait_asynccnt 4" ::: "memory"); }  // oldest slab landed
    else             { asm volatile("s_wait_asynccnt 0" ::: "memory"); }  // drain tail
    __syncthreads();                       // all waves landed + prev compute done
    if (ks + 2 < nk) {                     // keep 2 slabs in flight
      int nb = cur + 2; if (nb >= 3) nb -= 3;
      issue_tile(Ag, baseA[nb], (ks + 2) * 32);
      issue_tile(Bg, baseB[nb], (ks + 2) * 32);
    }
    const _Float16* la = lsA[cur];
    const _Float16* lb = lsB[cur];
    v16h fa0 = load_a_frag (la + (wm +  0) * LDS_STRIDE, LDS_STRIDE);
    v16h fa1 = load_a_frag (la + (wm + 16) * LDS_STRIDE, LDS_STRIDE);
    v16h fb0 = load_bt_frag(lb + (wn +  0) * LDS_STRIDE, LDS_STRIDE);
    v16h fb1 = load_bt_frag(lb + (wn + 16) * LDS_STRIDE, LDS_STRIDE);
    v16h fb2 = load_bt_frag(lb + (wn + 32) * LDS_STRIDE, LDS_STRIDE);
    v16h fb3 = load_bt_frag(lb + (wn + 48) * LDS_STRIDE, LDS_STRIDE);
    acc[0][0] = WMMA_F16(fa0, fb0, acc[0][0]);
    acc[0][1] = WMMA_F16(fa0, fb1, acc[0][1]);
    acc[0][2] = WMMA_F16(fa0, fb2, acc[0][2]);
    acc[0][3] = WMMA_F16(fa0, fb3, acc[0][3]);
    acc[1][0] = WMMA_F16(fa1, fb0, acc[1][0]);
    acc[1][1] = WMMA_F16(fa1, fb1, acc[1][1]);
    acc[1][2] = WMMA_F16(fa1, fb2, acc[1][2]);
    acc[1][3] = WMMA_F16(fa1, fb3, acc[1][3]);
    if (++cur == 3) cur = 0;
  }

#pragma unroll
  for (int i = 0; i < 2; ++i)
#pragma unroll
    for (int j = 0; j < 4; ++j) {
      const int mo = m0b + wm + i * 16;
      const int no = n0b + wn + j * 16;
      store_c_frag<OutT, GELU>(acc[i][j], C + (size_t)mo * N + no, N,
                               bias ? bias + no : nullptr, 1.0f);
    }
}

// ---------------- attention: scores = (Q Kh^T) * 1/sqrt(DH) ----------------
// grid (SS/128, SS/16, ZZ), block 256 (8 waves, each a 16x16 key tile)
__global__ void __launch_bounds__(256)
k_attn_scores(const _Float16* __restrict__ q, const _Float16* __restrict__ k,
              float* __restrict__ scores) {
  const int z  = blockIdx.z;                 // b*NHEAD + h
  const int b  = z / NHEAD, hh = z % NHEAD;
  const int wid = threadIdx.x >> 5;
  const int qi0 = blockIdx.y * 16;
  const int ki0 = blockIdx.x * 128 + wid * 16;
  const _Float16* Ap = q + ((size_t)(b * SS + qi0)) * HH + hh * DH;   // [16 x 64], lda=HH
  const _Float16* Bp = k + ((size_t)(b * SS + ki0)) * HH + hh * DH;   // BT rows = keys
  v8f acc = {};
  acc = WMMA_F16(load_a_frag(Ap, HH),      load_bt_frag(Bp, HH),      acc);
  acc = WMMA_F16(load_a_frag(Ap + 32, HH), load_bt_frag(Bp + 32, HH), acc);
  float* Cp = scores + ((size_t)z * SS + qi0) * SS + ki0;
  store_c_frag<float, false>(acc, Cp, SS, nullptr, 0.125f);           // 1/sqrt(64)
}

// ---------------- attention: ctx = P * V  (via vT[H][BS]) ----------------
// grid (SS/16, ZZ), block 128 (4 waves, each a 16-wide d tile)
__global__ void __launch_bounds__(128)
k_attn_ctx(const _Float16* __restrict__ probs, const _Float16* __restrict__ vT,
           _Float16* __restrict__ ctx) {
  const int z  = blockIdx.y;
  const int b  = z / NHEAD, hh = z % NHEAD;
  const int qi0 = blockIdx.x * 16;
  const int n0  = (threadIdx.x >> 5) * 16;   // d offset within head
  const _Float16* Ap = probs + ((size_t)z * SS + qi0) * SS;                 // lda=SS
  const _Float16* Bp = vT + (size_t)(hh * DH + n0) * BS + (size_t)b * SS;   // ldb=BS
  v8f acc = {};
  for (int k0 = 0; k0 < SS; k0 += 32)
    acc = WMMA_F16(load_a_frag(Ap + k0, SS), load_bt_frag(Bp + k0, BS), acc);
  _Float16* Cp = ctx + (size_t)(b * SS + qi0) * HH + hh * DH + n0;
  store_c_frag<_Float16, false>(acc, Cp, HH, nullptr, 1.0f);
}

// ---------------- tiled transpose (+cast) : in [K][N] -> out [N][K] f16 ----------------
template<typename InT>
__global__ void __launch_bounds__(256)
k_transpose(const InT* __restrict__ in, _Float16* __restrict__ out, int K, int N) {
  __shared__ _Float16 tile[32][33];
  const int k0 = blockIdx.y * 32, n0 = blockIdx.x * 32;
  const int tx = threadIdx.x & 31, ty = threadIdx.x >> 5;   // 8 rows per pass
#pragma unroll
  for (int r = ty; r < 32; r += 8)
    tile[r][tx] = (_Float16)in[(size_t)(k0 + r) * N + n0 + tx];
  __syncthreads();
#pragma unroll
  for (int r = ty; r < 32; r += 8)
    out[(size_t)(n0 + r) * K + k0 + tx] = tile[tx][r];
}

// ---------------- embeddings ----------------
__global__ void __launch_bounds__(256)
k_embed(const int* __restrict__ x, const float* __restrict__ we, const float* __restrict__ pe,
        const float* __restrict__ te, float* __restrict__ out) {
  const size_t row = blockIdx.x;             // b*SS + s
  const int s   = (int)(row & (SS - 1));
  const int tok = x[row];
#pragma unroll
  for (int j = 0; j < 3; ++j) {
    int i = threadIdx.x + j * 256;
    out[row * HH + i] = we[(size_t)tok * HH + i] + pe[(size_t)s * HH + i] + te[i];
  }
}

// ---------------- residual add + LayerNorm (writes f32 h and f16 h16) ----------------
__global__ void __launch_bounds__(256)
k_add_ln(const float* __restrict__ inp, const float* __restrict__ res,
         const float* __restrict__ g, const float* __restrict__ bta,
         float* __restrict__ out, _Float16* __restrict__ out16) {
  __shared__ float vb[HH];
  __shared__ float red[256];
  const int t = threadIdx.x;
  const size_t row = blockIdx.x;
  float s = 0.f;
#pragma unroll
  for (int j = 0; j < 3; ++j) {
    int i = t + j * 256;
    float v = inp[row * HH + i];
    if (res) v += res[row * HH + i];
    vb[i] = v; s += v;
  }
  red[t] = s; __syncthreads();
  for (int o = 128; o > 0; o >>= 1) { if (t < o) red[t] += red[t + o]; __syncthreads(); }
  const float mean = red[0] * (1.0f / HH);
  __syncthreads();
  s = 0.f;
#pragma unroll
  for (int j = 0; j < 3; ++j) { float d = vb[t + j * 256] - mean; s += d * d; }
  red[t] = s; __syncthreads();
  for (int o = 128; o > 0; o >>= 1) { if (t < o) red[t] += red[t + o]; __syncthreads(); }
  const float rstd = rsqrtf(red[0] * (1.0f / HH) + 1e-12f);
#pragma unroll
  for (int j = 0; j < 3; ++j) {
    int i = t + j * 256;
    float y = (vb[i] - mean) * rstd * g[i] + bta[i];
    out[row * HH + i] = y;
    if (out16) out16[row * HH + i] = (_Float16)y;
  }
}

// ---------------- softmax over last dim (256), emits f16 probs ----------------
__global__ void __launch_bounds__(256)
k_softmax(const float* __restrict__ sc, _Float16* __restrict__ pr) {
  __shared__ float red[256];
  const int t = threadIdx.x;
  const size_t row = blockIdx.x;
  const float v = sc[row * SS + t];
  red[t] = v; __syncthreads();
  for (int o = 128; o > 0; o >>= 1) { if (t < o) red[t] = fmaxf(red[t], red[t + o]); __syncthreads(); }
  const float m = red[0]; __syncthreads();
  const float e = __expf(v - m);
  red[t] = e; __syncthreads();
  for (int o = 128; o > 0; o >>= 1) { if (t < o) red[t] += red[t + o]; __syncthreads(); }
  pr[row * SS + t] = (_Float16)(e / red[0]);
}

// ---------------- classifier (N=9, tiny) ----------------
__global__ void __launch_bounds__(256)
k_cls(const float* __restrict__ h, const float* __restrict__ w,
      const float* __restrict__ b, float* __restrict__ em) {
  const int idx = blockIdx.x * 256 + threadIdx.x;
  if (idx >= BS * CC) return;
  const int c = idx % CC; const int r = idx / CC;
  float s = b[c];
  const float* hr = h + (size_t)r * HH;
  for (int kk = 0; kk < HH; ++kk) s += hr[kk] * w[(size_t)kk * CC + c];
  em[idx] = s;
}

// ---------------- CRF negative mean log-likelihood ----------------
__global__ void __launch_bounds__(32)
k_crf(const float* __restrict__ em, const int* __restrict__ tgt,
      const float* __restrict__ st, const float* __restrict__ en,
      const float* __restrict__ tr, float* __restrict__ out) {
  __shared__ float acc[BB];
  const int b = threadIdx.x;
  if (b < BB) {
    const float* e = em + (size_t)b * SS * CC;
    const int* t = tgt + (size_t)b * SS;
    float num = st[t[0]] + en[t[SS - 1]];
    for (int s2 = 0; s2 < SS; ++s2) num += e[s2 * CC + t[s2]];
    for (int s2 = 0; s2 < SS - 1; ++s2) num += tr[t[s2] * CC + t[s2 + 1]];
    float al[CC];
    for (int j = 0; j < CC; ++j) al[j] = st[j] + e[j];
    for (int s2 = 1; s2 < SS; ++s2) {
      float na[CC];
      for (int j = 0; j < CC; ++j) {
        float m = -1e30f;
        for (int i = 0; i < CC; ++i) m = fmaxf(m, al[i] + tr[i * CC + j]);
        float sum = 0.f;
        for (int i = 0; i < CC; ++i) sum += __expf(al[i] + tr[i * CC + j] - m);
        na[j] = m + __logf(sum) + e[s2 * CC + j];
      }
      for (int j = 0; j < CC; ++j) al[j] = na[j];
    }
    float m = -1e30f;
    for (int j = 0; j < CC; ++j) m = fmaxf(m, al[j] + en[j]);
    float sum = 0.f;
    for (int j = 0; j < CC; ++j) sum += __expf(al[j] + en[j] - m);
    acc[b] = num - (m + __logf(sum));
  }
  __syncthreads();
  if (threadIdx.x == 0) {
    float s = 0.f;
    for (int i = 0; i < BB; ++i) s += acc[i];
    out[0] = -s / (float)BB;
  }
}

// ---------------- host orchestration ----------------
struct LayerP {
  const float *q_w,*q_b,*k_w,*k_b,*v_w,*v_b,*o_w,*o_b,*ln1_g,*ln1_b,*ff1_w,*ff1_b,*ff2_w,*ff2_b,*ln2_g,*ln2_b;
};
struct ModelP {
  const float *cls_w,*cls_b,*crf_start,*crf_end,*crf_trans,*emb_ln_g,*emb_ln_b,*pos,*type,*word;
  const int *x,*target;
  LayerP L[LL];
};

extern "C" void kernel_launch(void* const* d_in, const int* in_sizes, int n_in,
                              void* d_out, int out_size, void* d_ws, size_t ws_size,
                              hipStream_t stream) {
  (void)out_size; (void)ws_size;
  ModelP P;
  // Detect pytree flatten order. JAX sorts dict keys (in_sizes[0] == CC for cls_b);
  // fall back to insertion order (in_sizes[0] == BS for int x).
  const bool sorted_order = !(n_in > 0 && in_sizes[0] == BS);
  auto F = [&](int i) { return (const float*)d_in[i]; };
  if (sorted_order) {
    P.cls_b = F(0); P.cls_w = F(1); P.crf_end = F(2); P.crf_start = F(3); P.crf_trans = F(4);
    P.emb_ln_b = F(5); P.emb_ln_g = F(6);
    for (int l = 0; l < LL; ++l) {
      const int o = 7 + l * 16;  // sorted: ff1_b,ff1_w,ff2_b,ff2_w,k_b,k_w,ln1_b,ln1_g,ln2_b,ln2_g,o_b,o_w,q_b,q_w,v_b,v_w
      P.L[l].ff1_b = F(o+0);  P.L[l].ff1_w = F(o+1);  P.L[l].ff2_b = F(o+2);  P.L[l].ff2_w = F(o+3);
      P.L[l].k_b   = F(o+4);  P.L[l].k_w   = F(o+5);  P.L[l].ln1_b = F(o+6);  P.L[l].ln1_g = F(o+7);
      P.L[l].ln2_b = F(o+8);  P.L[l].ln2_g = F(o+9);  P.L[l].o_b   = F(o+10); P.L[l].o_w   = F(o+11);
      P.L[l].q_b   = F(o+12); P.L[l].q_w   = F(o+13); P.L[l].v_b   = F(o+14); P.L[l].v_w   = F(o+15);
    }
    P.pos = F(199); P.type = F(200); P.word = F(201);
    P.target = (const int*)d_in[202]; P.x = (const int*)d_in[203];
  } else {
    P.x = (const int*)d_in[0]; P.target = (const int*)d_in[1];
    P.word = F(2); P.pos = F(3); P.type = F(4); P.emb_ln_g = F(5); P.emb_ln_b = F(6);
    for (int l = 0; l < LL; ++l) {
      const int o = 7 + l * 16;
      P.L[l].q_w = F(o+0);  P.L[l].q_b = F(o+1);  P.L[l].k_w = F(o+2);  P.L[l].k_b = F(o+3);
      P.L[l].v_w = F(o+4);  P.L[l].v_b = F(o+5);  P.L[l].o_w = F(o+6);  P.L[l].o_b = F(o+7);
      P.L[l].ln1_g = F(o+8); P.L[l].ln1_b = F(o+9); P.L[l].ff1_w = F(o+10); P.L[l].ff1_b = F(o+11);
      P.L[l].ff2_w = F(o+12); P.L[l].ff2_b = F(o+13); P.L[l].ln2_g = F(o+14); P.L[l].ln2_b = F(o+15);
    }
    P.cls_w = F(199); P.cls_b = F(200); P.crf_start = F(201); P.crf_end = F(202); P.crf_trans = F(203);
  }

  // ---- workspace carve-up ----
  char* ws = (char*)d_ws;
  size_t off = 0;
  auto carve = [&](size_t bytes) { void* p = ws + off; off = (off + bytes + 255) & ~(size_t)255; return p; };
  float*     h      = (float*)    carve((size_t)BS * HH * 4);
  float*     tmp    = (float*)    carve((size_t)BS * HH * 4);
  _Float16*  h16    = (_Float16*) carve((size_t)BS * HH * 2);
  _Float16*  q16    = (_Float16*) carve((size_t)BS * HH * 2);
  _Float16*  k16    = (_Float16*) carve((size_t)BS * HH * 2);
  _Float16*  v16    = (_Float16*) carve((size_t)BS * HH * 2);
  _Float16*  vT     = (_Float16*) carve((size_t)BS * HH * 2);
  _Float16*  ctx16  = (_Float16*) carve((size_t)BS * HH * 2);
  _Float16*  wT     = (_Float16*) carve((size_t)HH * FFD * 2);
  float*     scores = (float*)    carve((size_t)ZZ * SS * SS * 4);   // aliased with ff16
  _Float16*  ff16   = (_Float16*) scores;                            // FFN intermediate (disjoint lifetime)
  _Float16*  probs  = (_Float16*) carve((size_t)ZZ * SS * SS * 2);
  float*     em     = (float*)    carve((size_t)BS * CC * 4);

  const dim3 blk256(256), blk128(128);
  const dim3 g_gemm_768(HH / 128, BS / 128);     // N=768
  const dim3 g_gemm_ff(FFD / 128, BS / 128);     // N=3072
  const dim3 g_tr_hh(HH / 32, HH / 32);          // 768x768 weight transpose
  const dim3 g_tr_hf(FFD / 32, HH / 32);         // in [768][3072]
  const dim3 g_tr_fh(HH / 32, FFD / 32);         // in [3072][768]
  const dim3 g_tr_v(HH / 32, BS / 32);           // in [4096][768]

  // embeddings + LN
  k_embed<<<BS, blk256, 0, stream>>>(P.x, P.word, P.pos, P.type, tmp);
  k_add_ln<<<BS, blk256, 0, stream>>>(tmp, nullptr, P.emb_ln_g, P.emb_ln_b, h, h16);

  for (int l = 0; l < LL; ++l) {
    const LayerP& Ly = P.L[l];
    // Q/K/V projections (f16 out for WMMA attention)
    k_transpose<float><<<g_tr_hh, blk256, 0, stream>>>(Ly.q_w, wT, HH, HH);
    k_gemm_tn<_Float16, false><<<g_gemm_768, blk256, 0, stream>>>(h16, wT, Ly.q_b, q16, BS, HH, HH);
    k_transpose<float><<<g_tr_hh, blk256, 0, stream>>>(Ly.k_w, wT, HH, HH);
    k_gemm_tn<_Float16, false><<<g_gemm_768, blk256, 0, stream>>>(h16, wT, Ly.k_b, k16, BS, HH, HH);
    k_transpose<float><<<g_tr_hh, blk256, 0, stream>>>(Ly.v_w, wT, HH, HH);
    k_gemm_tn<_Float16, false><<<g_gemm_768, blk256, 0, stream>>>(h16, wT, Ly.v_b, v16, BS, HH, HH);
    // attention
    k_transpose<_Float16><<<g_tr_v, blk256, 0, stream>>>(v16, vT, BS, HH);
    k_attn_scores<<<dim3(SS / 128, SS / 16, ZZ), blk256, 0, stream>>>(q16, k16, scores);
    k_softmax<<<ZZ * SS, blk256, 0, stream>>>(scores, probs);
    k_attn_ctx<<<dim3(SS / 16, ZZ), blk128, 0, stream>>>(probs, vT, ctx16);
    // output projection + residual LN
    k_transpose<float><<<g_tr_hh, blk256, 0, stream>>>(Ly.o_w, wT, HH, HH);
    k_gemm_tn<float, false><<<g_gemm_768, blk256, 0, stream>>>(ctx16, wT, Ly.o_b, tmp, BS, HH, HH);
    k_add_ln<<<BS, blk256, 0, stream>>>(tmp, h, Ly.ln1_g, Ly.ln1_b, h, h16);
    // FFN: GEMM+bias+GELU fused (f16 out), then GEMM+bias, residual LN
    k_transpose<float><<<g_tr_hf, blk256, 0, stream>>>(Ly.ff1_w, wT, HH, FFD);
    k_gemm_tn<_Float16, true><<<g_gemm_ff, blk256, 0, stream>>>(h16, wT, Ly.ff1_b, ff16, BS, FFD, HH);
    k_transpose<float><<<g_tr_fh, blk256, 0, stream>>>(Ly.ff2_w, wT, FFD, HH);
    k_gemm_tn<float, false><<<g_gemm_768, blk256, 0, stream>>>(ff16, wT, Ly.ff2_b, tmp, BS, HH, FFD);
    k_add_ln<<<BS, blk256, 0, stream>>>(tmp, h, Ly.ln2_g, Ly.ln2_b, h, h16);
  }

  // classifier + CRF
  k_cls<<<(BS * CC + 255) / 256, blk256, 0, stream>>>(h, P.cls_w, P.cls_b, em);
  k_crf<<<1, 32, 0, stream>>>(em, P.target, P.crf_start, P.crf_end, P.crf_trans, (float*)d_out);
}
